// DCRNN_2671469658628
// MI455X (gfx1250) — matrix operand order repaired and checked by reading, not compiled, not run on hardware
//
#include <hip/hip_runtime.h>
#include <math.h>

#define NN 50000
#define EE 800000
#define HH 64
#define TT 12
#define PP 12
#define INF 3
#define OUTF 2

typedef __attribute__((ext_vector_type(16))) __bf16 v16bf;
typedef __attribute__((ext_vector_type(8)))  float v8f;

__device__ __forceinline__ unsigned short f32_to_bf16(float f) {
    unsigned int u = __float_as_uint(f);
    u += 0x7FFFu + ((u >> 16) & 1u);   // round-to-nearest-even
    return (unsigned short)(u >> 16);
}

// ---------------- utility kernels ----------------

__global__ void fill_zero_kernel(float* __restrict__ p, long long n) {
    long long i = (long long)blockIdx.x * blockDim.x + threadIdx.x;
    if (i < n) p[i] = 0.0f;
}

__global__ void deg_kernel(const int* __restrict__ dst, float* __restrict__ deg) {
    int e = blockIdx.x * blockDim.x + threadIdx.x;
    if (e < EE) atomicAdd(&deg[dst[e]], 1.0f);
}

__global__ void invdeg_kernel(float* __restrict__ deg) {
    int n = blockIdx.x * blockDim.x + threadIdx.x;
    if (n < NN) deg[n] = 1.0f / fmaxf(deg[n], 1.0f);
}

// W [Fin][Fout] fp32  ->  Wt [Fout][Kp] bf16 (transposed, zero-padded K)
__global__ void convert_wt_kernel(const float* __restrict__ W, unsigned short* __restrict__ Wt,
                                  int Fin, int Fout, int Kp) {
    int o = blockIdx.x;
    for (int k = threadIdx.x; k < Kp; k += blockDim.x) {
        float v = (k < Fin) ? W[(long long)k * Fout + o] : 0.0f;
        Wt[(long long)o * Kp + k] = f32_to_bf16(v);
    }
}

// X[n][k] = concat(A[n][0:Fa], (R? r*s : s))   zero-padded to Kp, bf16
__global__ void pack_kernel(const float* __restrict__ A, int aStride, int Fa,
                            const float* __restrict__ S, const float* __restrict__ R,
                            unsigned short* __restrict__ X, int Kp) {
    int n = blockIdx.x;
    for (int k = threadIdx.x; k < Kp; k += blockDim.x) {
        float v = 0.0f;
        if (k < Fa) {
            v = A[(long long)n * aStride + k];
        } else if (k < Fa + HH) {
            int h = k - Fa;
            v = S[(long long)n * HH + h];
            if (R) v *= R[(long long)n * (2 * HH) + h];   // r-gate (stride 2H)
        }
        X[(long long)n * Kp + k] = f32_to_bf16(v);
    }
}

// ---------------- WMMA bf16 GEMM: out[M][Fout] = X[M][Kp] * Wt[Fout][Kp]^T ----------------
// One wave per 16x16 C-tile; K stepped by 32 via v_wmma_f32_16x16x32_bf16.
__global__ void gemm_bf16_kernel(const unsigned short* __restrict__ X,
                                 const unsigned short* __restrict__ Wt,
                                 float* __restrict__ out,
                                 int M, int Kp, int Fout) {
    int wave = (int)(((long long)blockIdx.x * blockDim.x + threadIdx.x) >> 5);
    int lane = threadIdx.x & 31;
    int tilesN = Fout >> 4;
    int totalTiles = (M >> 4) * tilesN;
    if (wave >= totalTiles) return;           // wave-uniform: EXEC stays all-1 for WMMA

    int tm = wave / tilesN;
    int tn = wave - tm * tilesN;
    int half = lane >> 4;
    int mrow = (tm << 4) + (lane & 15);       // A row for this lane
    int col  = (tn << 4) + (lane & 15);       // B/C column for this lane

    union Frag { v16bf v; unsigned int u[8]; };
    Frag a, b;
    v8f acc = {0.f, 0.f, 0.f, 0.f, 0.f, 0.f, 0.f, 0.f};

    const unsigned short* arow = X  + (long long)mrow * Kp;
    const unsigned short* brow = Wt + (long long)col  * Kp;

    for (int k0 = 0; k0 < Kp; k0 += 32) {
        #pragma unroll
        for (int v = 0; v < 8; ++v) {
            // A 16x32 bf16 layout: VGPR v holds K = (v&3)*2 + (v>=4?16:0) + half*8 (pair)
            int ka = ((v & 3) << 1) + ((v >> 2) << 4) + (half << 3);
            a.u[v] = *(const unsigned int*)(arow + k0 + ka);
            // B 32x16 bf16 layout: VGPR v holds K = (v&3)*2 + (v>=4?16:0) + half*8 (pair), column = lane&15
            int kb = ((v & 3) << 1) + ((v >> 2) << 4) + (half << 3);
            b.u[v] = *(const unsigned int*)(brow + k0 + kb);
        }
        acc = __builtin_amdgcn_wmma_f32_16x16x32_bf16(false, a.v, false, b.v,
                                                      (short)0, acc, false, false);
    }
    // D layout: VGPR v -> row (v + half*8), col = lane&15
    #pragma unroll
    for (int v = 0; v < 8; ++v) {
        int rr = (tm << 4) + v + (half << 3);
        out[(long long)rr * Fout + col] = acc[v];
    }
}

// agg[dst[e]][:] += XW[src[e]][:]   (float4 per thread, L2-resident atomics)
__global__ void scatter_add_kernel(const float* __restrict__ XW, const int* __restrict__ src,
                                   const int* __restrict__ dst, float* __restrict__ agg,
                                   int lgF4) {
    long long t = (long long)blockIdx.x * blockDim.x + threadIdx.x;
    long long total = (long long)EE << lgF4;
    if (t >= total) return;
    int e = (int)(t >> lgF4);
    int j = ((int)t & ((1 << lgF4) - 1)) << 2;
    int F = 4 << lgF4;
    int s = src[e], d = dst[e];
    const float4 v = *(const float4*)(XW + (long long)s * F + j);
    float* o = agg + (long long)d * F + j;
    atomicAdd(o + 0, v.x); atomicAdd(o + 1, v.y);
    atomicAdd(o + 2, v.z); atomicAdd(o + 3, v.w);
}

// ru = sigmoid(agg * inv_deg + b)   [N,128]
__global__ void ru_final_kernel(const float* __restrict__ agg, const float* __restrict__ inv,
                                const float* __restrict__ b, float* __restrict__ ru) {
    int n = blockIdx.x, h = threadIdx.x;   // blockDim = 128
    float x = agg[(long long)n * 128 + h] * inv[n] + b[h];
    ru[(long long)n * 128 + h] = 1.0f / (1.0f + expf(-x));
}

// c = tanh(agg*inv+b); S = u*S + (1-u)*c   [N,64], u = ru[:,64:]
__global__ void cell_final_kernel(const float* __restrict__ agg, const float* __restrict__ inv,
                                  const float* __restrict__ b, const float* __restrict__ ru,
                                  float* __restrict__ S) {
    int n = blockIdx.x, h = threadIdx.x;   // blockDim = 64
    float c = tanhf(agg[(long long)n * 64 + h] * inv[n] + b[h]);
    float u = ru[(long long)n * 128 + 64 + h];
    float s = S[(long long)n * 64 + h];
    S[(long long)n * 64 + h] = u * s + (1.0f - u) * c;
}

// o = S1 @ out_W + out_b ; also writes d_out[:, p, :]
__global__ void outproj_kernel(const float* __restrict__ S1, const float* __restrict__ W,
                               const float* __restrict__ b, float* __restrict__ O,
                               float* __restrict__ dout, int p) {
    int n = blockIdx.x * blockDim.x + threadIdx.x;
    if (n >= NN) return;
    float a0 = b[0], a1 = b[1];
    const float* s = S1 + (long long)n * HH;
    #pragma unroll
    for (int h = 0; h < HH; ++h) { float sv = s[h]; a0 += sv * W[2 * h]; a1 += sv * W[2 * h + 1]; }
    O[2 * n] = a0; O[2 * n + 1] = a1;
    dout[(long long)n * PP * OUTF + p * OUTF + 0] = a0;
    dout[(long long)n * PP * OUTF + p * OUTF + 1] = a1;
}

// ---------------- host-side cell driver ----------------

static void run_cell(const float* inpA, int aStride, int Fa, int Kp,
                     float* S,
                     const unsigned short* WtRu, const float* bRu,
                     const unsigned short* WtC,  const float* bC,
                     unsigned short* X, float* XW, float* AGG, float* RU,
                     const float* INV, const int* src, const int* dst,
                     hipStream_t stream) {
    // ---- ru gconv (F_out = 128) ----
    pack_kernel<<<NN, 128, 0, stream>>>(inpA, aStride, Fa, S, nullptr, X, Kp);
    {
        int tiles = (NN / 16) * (128 / 16);
        gemm_bf16_kernel<<<(tiles + 7) / 8, 256, 0, stream>>>(X, WtRu, XW, NN, Kp, 128);
    }
    fill_zero_kernel<<<(int)(((long long)NN * 128 + 255) / 256), 256, 0, stream>>>(AGG, (long long)NN * 128);
    {
        long long total = (long long)EE * 32;
        scatter_add_kernel<<<(int)((total + 255) / 256), 256, 0, stream>>>(XW, src, dst, AGG, 5);
    }
    ru_final_kernel<<<NN, 128, 0, stream>>>(AGG, INV, bRu, RU);

    // ---- c gconv (F_out = 64), input uses r*S ----
    pack_kernel<<<NN, 128, 0, stream>>>(inpA, aStride, Fa, S, RU, X, Kp);
    {
        int tiles = (NN / 16) * (64 / 16);
        gemm_bf16_kernel<<<(tiles + 7) / 8, 256, 0, stream>>>(X, WtC, XW, NN, Kp, 64);
    }
    fill_zero_kernel<<<(int)(((long long)NN * 64 + 255) / 256), 256, 0, stream>>>(AGG, (long long)NN * 64);
    {
        long long total = (long long)EE * 16;
        scatter_add_kernel<<<(int)((total + 255) / 256), 256, 0, stream>>>(XW, src, dst, AGG, 4);
    }
    cell_final_kernel<<<NN, 64, 0, stream>>>(AGG, INV, bC, RU, S);
}

extern "C" void kernel_launch(void* const* d_in, const int* in_sizes, int n_in,
                              void* d_out, int out_size, void* d_ws, size_t ws_size,
                              hipStream_t stream) {
    (void)in_sizes; (void)n_in; (void)out_size; (void)ws_size;

    const float* x         = (const float*)d_in[0];
    const int*   src       = (const int*)d_in[1];
    const int*   dst       = (const int*)d_in[2];
    const float* enc_ru_W0 = (const float*)d_in[3];  const float* enc_ru_b0 = (const float*)d_in[4];
    const float* enc_c_W0  = (const float*)d_in[5];  const float* enc_c_b0  = (const float*)d_in[6];
    const float* enc_ru_W1 = (const float*)d_in[7];  const float* enc_ru_b1 = (const float*)d_in[8];
    const float* enc_c_W1  = (const float*)d_in[9];  const float* enc_c_b1  = (const float*)d_in[10];
    const float* dec_ru_W0 = (const float*)d_in[11]; const float* dec_ru_b0 = (const float*)d_in[12];
    const float* dec_c_W0  = (const float*)d_in[13]; const float* dec_c_b0  = (const float*)d_in[14];
    const float* dec_ru_W1 = (const float*)d_in[15]; const float* dec_ru_b1 = (const float*)d_in[16];
    const float* dec_c_W1  = (const float*)d_in[17]; const float* dec_c_b1  = (const float*)d_in[18];
    const float* out_W     = (const float*)d_in[19]; const float* out_b     = (const float*)d_in[20];
    float* dout = (float*)d_out;

    // ---- workspace carve-out ----
    char* ws = (char*)d_ws;
    size_t off = 0;
    auto take = [&](size_t bytes) -> char* {
        char* p = ws + off;
        off = (off + bytes + 255) & ~(size_t)255;
        return p;
    };
    float*          INV      = (float*)take((size_t)NN * 4);
    unsigned short* WtEncRu0 = (unsigned short*)take(128 * 96 * 2);
    unsigned short* WtEncC0  = (unsigned short*)take(64 * 96 * 2);
    unsigned short* WtEncRu1 = (unsigned short*)take(128 * 128 * 2);
    unsigned short* WtEncC1  = (unsigned short*)take(64 * 128 * 2);
    unsigned short* WtDecRu0 = (unsigned short*)take(128 * 96 * 2);
    unsigned short* WtDecC0  = (unsigned short*)take(64 * 96 * 2);
    unsigned short* WtDecRu1 = (unsigned short*)take(128 * 128 * 2);
    unsigned short* WtDecC1  = (unsigned short*)take(64 * 128 * 2);
    unsigned short* X   = (unsigned short*)take((size_t)NN * 128 * 2);
    float* XW  = (float*)take((size_t)NN * 128 * 4);
    float* AGG = (float*)take((size_t)NN * 128 * 4);
    float* RU  = (float*)take((size_t)NN * 128 * 4);
    float* S0  = (float*)take((size_t)NN * 64 * 4);
    float* S1  = (float*)take((size_t)NN * 64 * 4);
    float* O   = (float*)take((size_t)NN * 2 * 4);

    // ---- one-time setup (runs every call; deterministic) ----
    fill_zero_kernel<<<(NN + 255) / 256, 256, 0, stream>>>(INV, NN);
    deg_kernel<<<(EE + 255) / 256, 256, 0, stream>>>(dst, INV);
    invdeg_kernel<<<(NN + 255) / 256, 256, 0, stream>>>(INV);

    convert_wt_kernel<<<128, 128, 0, stream>>>(enc_ru_W0, WtEncRu0, INF + HH, 128, 96);
    convert_wt_kernel<<< 64, 128, 0, stream>>>(enc_c_W0,  WtEncC0,  INF + HH,  64, 96);
    convert_wt_kernel<<<128, 128, 0, stream>>>(enc_ru_W1, WtEncRu1, 2 * HH,   128, 128);
    convert_wt_kernel<<< 64, 128, 0, stream>>>(enc_c_W1,  WtEncC1,  2 * HH,    64, 128);
    convert_wt_kernel<<<128, 128, 0, stream>>>(dec_ru_W0, WtDecRu0, OUTF + HH, 128, 96);
    convert_wt_kernel<<< 64, 128, 0, stream>>>(dec_c_W0,  WtDecC0,  OUTF + HH,  64, 96);
    convert_wt_kernel<<<128, 128, 0, stream>>>(dec_ru_W1, WtDecRu1, 2 * HH,   128, 128);
    convert_wt_kernel<<< 64, 128, 0, stream>>>(dec_c_W1,  WtDecC1,  2 * HH,    64, 128);

    fill_zero_kernel<<<(int)(((long long)NN * 64 + 255) / 256), 256, 0, stream>>>(S0, (long long)NN * 64);
    fill_zero_kernel<<<(int)(((long long)NN * 64 + 255) / 256), 256, 0, stream>>>(S1, (long long)NN * 64);

    // ---- encoder: time-major interleave of the two stacked layers ----
    for (int t = 0; t < TT; ++t) {
        run_cell(x + (size_t)t * INF, TT * INF, INF, 96, S0,
                 WtEncRu0, enc_ru_b0, WtEncC0, enc_c_b0,
                 X, XW, AGG, RU, INV, src, dst, stream);
        run_cell(S0, HH, HH, 128, S1,
                 WtEncRu1, enc_ru_b1, WtEncC1, enc_c_b1,
                 X, XW, AGG, RU, INV, src, dst, stream);
    }

    // ---- decoder: autoregressive, states continue from encoder finals ----
    fill_zero_kernel<<<(int)(((long long)NN * 2 + 255) / 256), 256, 0, stream>>>(O, (long long)NN * 2);
    for (int p = 0; p < PP; ++p) {
        run_cell(O, OUTF, OUTF, 96, S0,
                 WtDecRu0, dec_ru_b0, WtDecC0, dec_c_b0,
                 X, XW, AGG, RU, INV, src, dst, stream);
        run_cell(S0, HH, HH, 128, S1,
                 WtDecRu1, dec_ru_b1, WtDecC1, dec_c_b1,
                 X, XW, AGG, RU, INV, src, dst, stream);
        outproj_kernel<<<(NN + 255) / 256, 256, 0, stream>>>(S1, out_W, out_b, O, dout, p);
    }
}
